// ConvAttention_50611894616395
// MI455X (gfx1250) — compile-verified, matrix operand order checked by reading, not run on hardware
//
#include <hip/hip_runtime.h>
#include <math.h>

typedef __attribute__((ext_vector_type(16))) _Float16 v16h;
typedef __attribute__((ext_vector_type(8)))  float    v8f;

#define B_     8
#define C_     384
#define H_     32
#define W_     32
#define N_     1024
#define HEADS_ 8
#define HD_    48
#define HDP_   64

// ---------------------------------------------------------------- helpers

__device__ __forceinline__ v8f zero8() {
  v8f z;
#pragma unroll
  for (int i = 0; i < 8; ++i) z[i] = 0.0f;
  return z;
}

__device__ __forceinline__ v8f wmma32(v16h a, v16h b, v8f c) {
  // D = A(16x32 f16) * B(32x16 f16) + C(16x16 f32)
  return __builtin_amdgcn_wmma_f32_16x16x32_f16(
      /*neg_a=*/false, a, /*neg_b=*/false, b,
      /*c_mod=*/(short)0, c, /*reuse_a=*/false, /*reuse_b=*/false);
}

// Fragment-major LDS: frag[32 lanes][16 halves]; lane's 16 halves contiguous.
// ISA 7.12.2 16-bit fragment: lane covers (row = lane&15), K-halves
//   halfidx 0..7  -> K = kmid + 0..7        (kmid = (lane&16) ? 8 : 0)
//   halfidx 8..15 -> K = 16 + kmid + 0..7
// So per (lane, group g in {0,1}) the 8 halves are 8 CONSECUTIVE K values
// starting at kbase = g*16 + kmid  -> stage with one b128 copy.
__device__ __forceinline__ v16h ld_frag(const _Float16* p) {
  return *(const v16h*)p;  // 32B, two ds_load_b128
}

// ---------------------------------------------------------------- stage 1: dwconv + stats

__global__ __launch_bounds__(256) void dwconv_stats_kernel(
    const float* __restrict__ x,
    const float* __restrict__ dwq, const float* __restrict__ dwk,
    const float* __restrict__ dwv,
    float* __restrict__ y3, float* __restrict__ ssum, float* __restrict__ ssq)
{
  const int ch = blockIdx.x, b = blockIdx.y, br = blockIdx.z;
  const float* dw = (br == 0) ? dwq : (br == 1) ? dwk : dwv;
  float w[9];
#pragma unroll
  for (int i = 0; i < 9; ++i) w[i] = dw[ch * 9 + i];
  const float* xp = x + ((size_t)b * C_ + ch) * N_;
  float* yp = y3 + (((size_t)br * B_ + b) * C_ + ch) * N_;
  float s = 0.0f, s2 = 0.0f;
  for (int e = threadIdx.x; e < N_; e += 256) {
    int r = e >> 5, c = e & 31;
    float acc = 0.0f;
#pragma unroll
    for (int kr = 0; kr < 3; ++kr) {
      int rr = r + kr - 1;
      if (rr < 0 || rr >= H_) continue;
#pragma unroll
      for (int kc = 0; kc < 3; ++kc) {
        int cc = c + kc - 1;
        if (cc < 0 || cc >= W_) continue;
        acc += w[kr * 3 + kc] * xp[rr * W_ + cc];
      }
    }
    yp[e] = acc;
    s += acc;
    s2 += acc * acc;
  }
  __shared__ float rs[256], rs2[256];
  rs[threadIdx.x] = s; rs2[threadIdx.x] = s2;
  __syncthreads();
  for (int o = 128; o > 0; o >>= 1) {
    if ((int)threadIdx.x < o) {
      rs[threadIdx.x] += rs[threadIdx.x + o];
      rs2[threadIdx.x] += rs2[threadIdx.x + o];
    }
    __syncthreads();
  }
  if (threadIdx.x == 0) {
    atomicAdd(&ssum[br * C_ + ch], rs[0]);
    atomicAdd(&ssq[br * C_ + ch], rs2[0]);
  }
}

__global__ void finalize_stats_kernel(const float* __restrict__ ssum,
                                      const float* __restrict__ ssq,
                                      float* __restrict__ mean,
                                      float* __restrict__ rstd, int n, float cnt)
{
  int i = blockIdx.x * blockDim.x + threadIdx.x;
  if (i < n) {
    float m = ssum[i] / cnt;
    float v = ssq[i] / cnt - m * m;
    mean[i] = m;
    rstd[i] = rsqrtf(v + 1e-5f);
  }
}

// ---------------------------------------------------------------- stage 2: BN + GELU -> f16 (b, n, c)

__global__ __launch_bounds__(256) void bngelu_pack_kernel(
    const float* __restrict__ y3, const float* __restrict__ mean,
    const float* __restrict__ rstd,
    const float* __restrict__ gq, const float* __restrict__ bq,
    const float* __restrict__ gk, const float* __restrict__ bk,
    const float* __restrict__ gv, const float* __restrict__ bv,
    _Float16* __restrict__ yh)
{
  size_t idx = (size_t)blockIdx.x * 256 + threadIdx.x;
  if (idx >= (size_t)3 * B_ * C_ * N_) return;
  int br = (int)(idx / ((size_t)B_ * C_ * N_));
  int b  = (int)((idx / ((size_t)C_ * N_)) % B_);
  int ch = (int)((idx / N_) % C_);
  int n  = (int)(idx % N_);
  const float* g = (br == 0) ? gq : (br == 1) ? gk : gv;
  const float* bb = (br == 0) ? bq : (br == 1) ? bk : bv;
  float val = (y3[idx] - mean[br * C_ + ch]) * rstd[br * C_ + ch] * g[ch] + bb[ch];
  val = 0.5f * val * (1.0f + erff(val * 0.70710678118654752f));  // exact GELU
  // (br, b, n, c) channel-contiguous for vectorized GEMM staging
  yh[((size_t)(br * B_ + b) * N_ + n) * C_ + ch] = (_Float16)val;
}

__global__ void pack_w_kernel(const float* __restrict__ pwq,
                              const float* __restrict__ pwk,
                              const float* __restrict__ pwv,
                              const float* __restrict__ wo,
                              _Float16* __restrict__ Wh)
{
  int idx = blockIdx.x * 256 + threadIdx.x;
  if (idx >= 4 * C_ * C_) return;
  int sel = idx / (C_ * C_), r = idx % (C_ * C_);
  const float* src = (sel == 0) ? pwq : (sel == 1) ? pwk : (sel == 2) ? pwv : wo;
  Wh[idx] = (_Float16)src[r];
}

// ---------------------------------------------------------------- GEMM staging helper
// Decompose thread id into one b128 staging op of a 4-fragment tile.
struct StageIdx { int frag, lane, g, row, kbase; };
__device__ __forceinline__ StageIdx stage_idx(int t) {
  StageIdx si;
  si.frag = t >> 6;
  int sub = t & 63;
  si.lane = sub >> 1;
  si.g = sub & 1;
  si.row = si.lane & 15;
  int kmid = (si.lane & 16) ? 8 : 0;
  si.kbase = si.g * 16 + kmid;
  return si;
}

// ---------------------------------------------------------------- stage 3: 1x1 conv GEMMs (WMMA)

// out[o][n] = sum_c W[o][c] * Y[n][c]; W row-major (o,c), Y (n,c). Tile 64x64.
__global__ __launch_bounds__(256) void gemm_qkv_kernel(
    const _Float16* __restrict__ Wh, const _Float16* __restrict__ yh,
    _Float16* __restrict__ qkvh)
{
  const int n0 = blockIdx.x * 64, m0 = blockIdx.y * 64;
  const int br = blockIdx.z / B_, b = blockIdx.z % B_;
  const _Float16* Wbr = Wh + (size_t)br * C_ * C_;
  const _Float16* Bmat = yh + (size_t)(br * B_ + b) * N_ * C_;
  __shared__ __align__(32) _Float16 Af[4][32][16];
  __shared__ __align__(32) _Float16 Bf[4][32][16];
  const int wave = threadIdx.x >> 5, lane = threadIdx.x & 31;
  const int wm = wave & 3, wn = wave >> 2;
  const StageIdx si = stage_idx(threadIdx.x);
  v8f acc0 = zero8(), acc1 = zero8();
  for (int kt = 0; kt < C_ / 32; ++kt) {
    __syncthreads();
    // A: W[(m0 + frag*16 + row)][kt*32 + kbase .. +7]  (8 halves = b128)
    *(uint4*)&Af[si.frag][si.lane][si.g * 8] =
        *(const uint4*)&Wbr[(size_t)(m0 + si.frag * 16 + si.row) * C_ + kt * 32 + si.kbase];
    // B: Y[(n0 + frag*16 + row)][kt*32 + kbase .. +7]
    *(uint4*)&Bf[si.frag][si.lane][si.g * 8] =
        *(const uint4*)&Bmat[(size_t)(n0 + si.frag * 16 + si.row) * C_ + kt * 32 + si.kbase];
    if (kt + 1 < C_ / 32)
      __builtin_prefetch(&Bmat[(size_t)(n0 + si.frag * 16 + si.row) * C_ + kt * 32 + 32], 0, 1);
    __syncthreads();
    v16h a  = ld_frag(&Af[wm][lane][0]);
    v16h b0 = ld_frag(&Bf[wn * 2][lane][0]);
    v16h b1 = ld_frag(&Bf[wn * 2 + 1][lane][0]);
    acc0 = wmma32(a, b0, acc0);
    acc1 = wmma32(a, b1, acc1);
  }
  const float qscale = (br == 0) ? 0.14433756729740643f : 1.0f;  // 1/sqrt(48) folded into q
  const int nlane = lane & 15, mb = (lane < 16) ? 0 : 8;
  _Float16* dst = qkvh + (size_t)(br * B_ + b) * HEADS_ * N_ * HDP_;
#pragma unroll
  for (int j = 0; j < 2; ++j) {
    v8f acc = j ? acc1 : acc0;
#pragma unroll
    for (int r = 0; r < 8; ++r) {
      int o = m0 + wm * 16 + mb + r;
      int n = n0 + wn * 32 + j * 16 + nlane;
      int h = o / HD_, c = o - h * HD_;  // 48 | 16*k -> tiles never straddle heads
      _Float16 v = (_Float16)(acc[r] * qscale);
      if (br == 2)  // V: (h, c, n) -- kv-contiguous for P*V^T B-fragments
        dst[((size_t)h * HDP_ + c) * N_ + n] = v;
      else          // Q,K: (h, n, c) -- c-contiguous for QK^T fragments
        dst[((size_t)h * N_ + n) * HDP_ + c] = v;
    }
  }
}

__global__ __launch_bounds__(256) void gemm_proj_kernel(
    const _Float16* __restrict__ Wo, const _Float16* __restrict__ atth,
    float* __restrict__ proj)
{
  const int n0 = blockIdx.x * 64, m0 = blockIdx.y * 64;
  const int b = blockIdx.z;
  const _Float16* Bmat = atth + (size_t)b * N_ * C_;
  __shared__ __align__(32) _Float16 Af[4][32][16];
  __shared__ __align__(32) _Float16 Bf[4][32][16];
  const int wave = threadIdx.x >> 5, lane = threadIdx.x & 31;
  const int wm = wave & 3, wn = wave >> 2;
  const StageIdx si = stage_idx(threadIdx.x);
  v8f acc0 = zero8(), acc1 = zero8();
  for (int kt = 0; kt < C_ / 32; ++kt) {
    __syncthreads();
    *(uint4*)&Af[si.frag][si.lane][si.g * 8] =
        *(const uint4*)&Wo[(size_t)(m0 + si.frag * 16 + si.row) * C_ + kt * 32 + si.kbase];
    *(uint4*)&Bf[si.frag][si.lane][si.g * 8] =
        *(const uint4*)&Bmat[(size_t)(n0 + si.frag * 16 + si.row) * C_ + kt * 32 + si.kbase];
    if (kt + 1 < C_ / 32)
      __builtin_prefetch(&Bmat[(size_t)(n0 + si.frag * 16 + si.row) * C_ + kt * 32 + 32], 0, 1);
    __syncthreads();
    v16h a  = ld_frag(&Af[wm][lane][0]);
    v16h b0 = ld_frag(&Bf[wn * 2][lane][0]);
    v16h b1 = ld_frag(&Bf[wn * 2 + 1][lane][0]);
    acc0 = wmma32(a, b0, acc0);
    acc1 = wmma32(a, b1, acc1);
  }
  const int nlane = lane & 15, mb = (lane < 16) ? 0 : 8;
#pragma unroll
  for (int j = 0; j < 2; ++j) {
    v8f acc = j ? acc1 : acc0;
#pragma unroll
    for (int r = 0; r < 8; ++r) {
      int o = m0 + wm * 16 + mb + r;
      int n = n0 + wn * 32 + j * 16 + nlane;
      proj[((size_t)b * C_ + o) * N_ + n] = acc[r];
    }
  }
}

// ---------------------------------------------------------------- stage 4: flash attention (WMMA)

// grid (N/128, B*heads); 8 waves, wave w owns query rows q0+16w..q0+16w+15
__global__ __launch_bounds__(256) void attention_kernel(
    const _Float16* __restrict__ qkvh, _Float16* __restrict__ atth)
{
  const int q0 = blockIdx.x * 128;
  const int bh = blockIdx.y;
  const int b = bh / HEADS_, h = bh % HEADS_;
  const size_t blk = (size_t)HEADS_ * N_ * HDP_;
  const _Float16* qb = qkvh + (size_t)(0 * B_ + b) * blk + (size_t)h * N_ * HDP_;  // (n, c)
  const _Float16* kb = qkvh + (size_t)(1 * B_ + b) * blk + (size_t)h * N_ * HDP_;  // (n, c)
  const _Float16* vb = qkvh + (size_t)(2 * B_ + b) * blk + (size_t)h * HDP_ * N_;  // (c, n)

  // fragment-major LDS: [frag][kstep][lane][16 halves]
  __shared__ __align__(32) _Float16 Qf[8][2][32][16];  // A-frags of QK^T (per wave)
  __shared__ __align__(32) _Float16 Kf[4][2][32][16];  // B-frags of QK^T (kv groups)
  __shared__ __align__(32) _Float16 Vf[4][2][32][16];  // B-frags of P*V^T (c groups)
  __shared__ __align__(32) _Float16 Pf[8][2][32][16];  // A-frags of P*V^T (per wave)

  const int wave = threadIdx.x >> 5, lane = threadIdx.x & 31;
  const int nlane = lane & 15, mb = (lane < 16) ? 0 : 8;

  // stage all Q fragments once: 1024 b128 ops, 4 per thread
#pragma unroll
  for (int i = 0; i < 4; ++i) {
    int o = threadIdx.x + i * 256;
    int wq = o >> 7, rem = o & 127;
    int ks = rem >> 6, sub = rem & 63;
    int ln = sub >> 1, g = sub & 1;
    int m = ln & 15, kmid = (ln & 16) ? 8 : 0;
    int kbase = ks * 32 + g * 16 + kmid;
    *(uint4*)&Qf[wq][ks][ln][g * 8] =
        *(const uint4*)&qb[(size_t)(q0 + wq * 16 + m) * HDP_ + kbase];
  }

  float Mrow[8], Lrow[8];
  v8f oacc[4];
#pragma unroll
  for (int r = 0; r < 8; ++r) { Mrow[r] = -1e30f; Lrow[r] = 0.0f; }
#pragma unroll
  for (int j = 0; j < 4; ++j) oacc[j] = zero8();

  for (int kt = 0; kt < N_ / 64; ++kt) {
    const int kv0 = kt * 64;
    __syncthreads();  // K/V tiles of previous iteration fully consumed
    // stage K fragments: 512 ops, 2 per thread
#pragma unroll
    for (int i = 0; i < 2; ++i) {
      int o = threadIdx.x + i * 256;
      int ng = o >> 7, rem = o & 127;
      int ks = rem >> 6, sub = rem & 63;
      int ln = sub >> 1, g = sub & 1;
      int n = ln & 15, kmid = (ln & 16) ? 8 : 0;
      int kbase = ks * 32 + g * 16 + kmid;
      *(uint4*)&Kf[ng][ks][ln][g * 8] =
          *(const uint4*)&kb[(size_t)(kv0 + ng * 16 + n) * HDP_ + kbase];
    }
    // stage V fragments: B[kv][c] -> per lane 8 consecutive kv at fixed c
#pragma unroll
    for (int i = 0; i < 2; ++i) {
      int o = threadIdx.x + i * 256;
      int cg = o >> 7, rem = o & 127;
      int ks = rem >> 6, sub = rem & 63;
      int ln = sub >> 1, g = sub & 1;
      int c = cg * 16 + (ln & 15), kvmid = (ln & 16) ? 8 : 0;
      int kbase = ks * 32 + g * 16 + kvmid;
      *(uint4*)&Vf[cg][ks][ln][g * 8] =
          *(const uint4*)&vb[(size_t)c * N_ + kv0 + kbase];
    }
    __syncthreads();

    // S(16x64) = Q_tile^T K_tile
    v16h aq0 = ld_frag(&Qf[wave][0][lane][0]);
    v16h aq1 = ld_frag(&Qf[wave][1][lane][0]);
    v8f s[4];
#pragma unroll
    for (int j = 0; j < 4; ++j) {
      v16h bk0 = ld_frag(&Kf[j][0][lane][0]);
      v16h bk1 = ld_frag(&Kf[j][1][lane][0]);
      s[j] = wmma32(aq0, bk0, zero8());
      s[j] = wmma32(aq1, bk1, s[j]);
    }

    // online softmax; row values live in one VGPR across a 16-lane half
#pragma unroll
    for (int r = 0; r < 8; ++r) {
      float mc = fmaxf(fmaxf(s[0][r], s[1][r]), fmaxf(s[2][r], s[3][r]));
#pragma unroll
      for (int msk = 8; msk >= 1; msk >>= 1) mc = fmaxf(mc, __shfl_xor(mc, msk, 32));
      float mn = fmaxf(Mrow[r], mc);
      float corr = expf(Mrow[r] - mn);
      float ls = 0.0f;
#pragma unroll
      for (int j = 0; j < 4; ++j) {
        float p = expf(s[j][r] - mn);
        s[j][r] = p;
        ls += p;
      }
#pragma unroll
      for (int msk = 8; msk >= 1; msk >>= 1) ls += __shfl_xor(ls, msk, 32);
      Lrow[r] = Lrow[r] * corr + ls;
      Mrow[r] = mn;
#pragma unroll
      for (int j = 0; j < 4; ++j) oacc[j][r] = oacc[j][r] * corr;
    }

    // write P directly into its A-fragment layout (own wave's region only;
    // same-wave DS ordering makes the following loads safe without a barrier)
    // element (m = mb+r, kv = j*16+nlane):
    //   ks = j>>1 ; lane' = m + ((nlane&8)?16:0) ; halfidx = (nlane&7) + ((j&1)?8:0)
#pragma unroll
    for (int j = 0; j < 4; ++j) {
      int ks = j >> 1, hod = (j & 1) ? 8 : 0;
      int lhalf = (nlane & 8) ? 16 : 0, hlo = nlane & 7;
#pragma unroll
      for (int r = 0; r < 8; ++r)
        Pf[wave][ks][mb + r + lhalf][hlo + hod] = (_Float16)s[j][r];
    }

    v16h pa0 = ld_frag(&Pf[wave][0][lane][0]);
    v16h pa1 = ld_frag(&Pf[wave][1][lane][0]);
#pragma unroll
    for (int j = 0; j < 4; ++j) {
      v16h bv0 = ld_frag(&Vf[j][0][lane][0]);
      v16h bv1 = ld_frag(&Vf[j][1][lane][0]);
      oacc[j] = wmma32(pa0, bv0, oacc[j]);
      oacc[j] = wmma32(pa1, bv1, oacc[j]);
    }
  }

  // O /= L ; write as (b, n, 48h+c) f16 for the out-projection GEMM
#pragma unroll
  for (int j = 0; j < 4; ++j) {
#pragma unroll
    for (int r = 0; r < 8; ++r) {
      int c = j * 16 + nlane;
      if (c < HD_) {
        int n = q0 + wave * 16 + mb + r;
        atth[((size_t)b * N_ + n) * C_ + h * HD_ + c] =
            (_Float16)(oacc[j][r] / Lrow[r]);
      }
    }
  }
}

// ---------------------------------------------------------------- stage 5: output BN

__global__ __launch_bounds__(256) void out_stats_kernel(
    const float* __restrict__ proj, float* __restrict__ ssum, float* __restrict__ ssq)
{
  const int ch = blockIdx.x, b = blockIdx.y;
  const float* p = proj + ((size_t)b * C_ + ch) * N_;
  float s = 0.0f, s2 = 0.0f;
  for (int e = threadIdx.x; e < N_; e += 256) {
    float v = p[e];
    s += v;
    s2 += v * v;
  }
  __shared__ float rs[256], rs2[256];
  rs[threadIdx.x] = s; rs2[threadIdx.x] = s2;
  __syncthreads();
  for (int o = 128; o > 0; o >>= 1) {
    if ((int)threadIdx.x < o) {
      rs[threadIdx.x] += rs[threadIdx.x + o];
      rs2[threadIdx.x] += rs2[threadIdx.x + o];
    }
    __syncthreads();
  }
  if (threadIdx.x == 0) {
    atomicAdd(&ssum[3 * C_ + ch], rs[0]);
    atomicAdd(&ssq[3 * C_ + ch], rs2[0]);
  }
}

__global__ __launch_bounds__(256) void bn_apply_kernel(
    const float* __restrict__ proj, const float* __restrict__ mean,
    const float* __restrict__ rstd, const float* __restrict__ g,
    const float* __restrict__ bb, float* __restrict__ out)
{
  size_t idx = (size_t)blockIdx.x * 256 + threadIdx.x;
  if (idx >= (size_t)B_ * C_ * N_) return;
  int ch = (int)((idx / N_) % C_);
  out[idx] = (proj[idx] - mean[ch]) * rstd[ch] * g[ch] + bb[ch];
}

// ---------------------------------------------------------------- launcher

extern "C" void kernel_launch(void* const* d_in, const int* in_sizes, int n_in,
                              void* d_out, int out_size, void* d_ws, size_t ws_size,
                              hipStream_t stream)
{
  (void)in_sizes; (void)n_in; (void)out_size; (void)ws_size;
  const float* x     = (const float*)d_in[0];
  const float* dw_q  = (const float*)d_in[1];
  const float* g_q   = (const float*)d_in[2];
  const float* b_q   = (const float*)d_in[3];
  const float* pw_q  = (const float*)d_in[4];
  const float* dw_k  = (const float*)d_in[5];
  const float* g_k   = (const float*)d_in[6];
  const float* b_k   = (const float*)d_in[7];
  const float* pw_k  = (const float*)d_in[8];
  const float* dw_v  = (const float*)d_in[9];
  const float* g_v   = (const float*)d_in[10];
  const float* b_v   = (const float*)d_in[11];
  const float* pw_v  = (const float*)d_in[12];
  const float* w_out = (const float*)d_in[13];
  const float* g_out = (const float*)d_in[14];
  const float* b_out = (const float*)d_in[15];
  float* out = (float*)d_out;

  char* ws = (char*)d_ws;
  size_t off = 0;
  auto take = [&](size_t bytes) -> char* {
    char* p = ws + off;
    off = (off + bytes + 255) & ~(size_t)255;
    return p;
  };
  float*    y3   = (float*)take((size_t)3 * B_ * C_ * N_ * 4);
  _Float16* yh   = (_Float16*)take((size_t)3 * B_ * C_ * N_ * 2);   // (br,b,n,c)
  _Float16* qkvh = (_Float16*)take((size_t)3 * B_ * HEADS_ * HDP_ * N_ * 2);
  _Float16* Wh   = (_Float16*)take((size_t)4 * C_ * C_ * 2);
  _Float16* atth = (_Float16*)take((size_t)B_ * C_ * N_ * 2);       // (b,n,c)
  float*    proj = (float*)take((size_t)B_ * C_ * N_ * 4);
  float*    ssum = (float*)take((size_t)4 * C_ * 4);
  float*    ssq  = (float*)take((size_t)4 * C_ * 4);
  float*    mean = (float*)take((size_t)4 * C_ * 4);
  float*    rstd = (float*)take((size_t)4 * C_ * 4);

  hipMemsetAsync(ssum, 0, (size_t)4 * C_ * 4, stream);
  hipMemsetAsync(ssq, 0, (size_t)4 * C_ * 4, stream);
  hipMemsetAsync(qkvh, 0, (size_t)3 * B_ * HEADS_ * HDP_ * N_ * 2, stream);  // zero head pad

  dwconv_stats_kernel<<<dim3(C_, B_, 3), 256, 0, stream>>>(x, dw_q, dw_k, dw_v, y3, ssum, ssq);
  finalize_stats_kernel<<<(3 * C_ + 255) / 256, 256, 0, stream>>>(
      ssum, ssq, mean, rstd, 3 * C_, (float)(B_ * N_));
  {
    size_t total = (size_t)3 * B_ * C_ * N_;
    bngelu_pack_kernel<<<(unsigned)((total + 255) / 256), 256, 0, stream>>>(
        y3, mean, rstd, g_q, b_q, g_k, b_k, g_v, b_v, yh);
  }
  pack_w_kernel<<<(4 * C_ * C_ + 255) / 256, 256, 0, stream>>>(pw_q, pw_k, pw_v, w_out, Wh);
  gemm_qkv_kernel<<<dim3(N_ / 64, C_ / 64, 3 * B_), 256, 0, stream>>>(Wh, yh, qkvh);
  attention_kernel<<<dim3(N_ / 128, B_ * HEADS_), 256, 0, stream>>>(qkvh, atth);
  gemm_proj_kernel<<<dim3(N_ / 64, C_ / 64, B_), 256, 0, stream>>>(
      Wh + (size_t)3 * C_ * C_, atth, proj);
  out_stats_kernel<<<dim3(C_, B_), 256, 0, stream>>>(proj, ssum, ssq);
  finalize_stats_kernel<<<(C_ + 255) / 256, 256, 0, stream>>>(
      ssum + 3 * C_, ssq + 3 * C_, mean + 3 * C_, rstd + 3 * C_, C_, (float)(B_ * N_));
  {
    size_t total = (size_t)B_ * C_ * N_;
    bn_apply_kernel<<<(unsigned)((total + 255) / 256), 256, 0, stream>>>(
        proj, mean + 3 * C_, rstd + 3 * C_, g_out, b_out, out);
  }
}
